// TCMamba_40097814675764
// MI455X (gfx1250) — compile-verified, matrix operand order checked by reading, not run on hardware
//
#include <hip/hip_runtime.h>
#include <hip/hip_bf16.h>

// ---------------------------------------------------------------------------
// Problem constants (match reference)
// ---------------------------------------------------------------------------
#define kB       4
#define kL       1024
#define kDMODEL  512
#define kDSTATE  16
#define kDINNER  1024
#define kDTRANK  32
#define kM       (kB * kL)        // 4096 token rows

typedef __attribute__((ext_vector_type(16))) _Float16 v16h;
typedef __attribute__((ext_vector_type(8)))  _Float16 v8h;
typedef __attribute__((ext_vector_type(4)))  _Float16 v4h;
typedef __attribute__((ext_vector_type(8)))  float    v8f;

// ---------------------------------------------------------------------------
// Device helpers
// ---------------------------------------------------------------------------
__device__ __forceinline__ float silu_f(float x) {
    return x / (1.0f + __expf(-x));
}
__device__ __forceinline__ float softplus_f(float x) {
    return (x > 20.0f) ? x : log1pf(__expf(x));
}

// ---------------------------------------------------------------------------
// WMMA GEMM:  C[M,N] = act( A[M,K] * W[N,K]^T + bias[N] )
//
// f32 in memory, f16 staged through double-buffered LDS, f32 accumulation via
// v_wmma_f32_16x16x32_f16.
//
// Block = 256 threads = 8 waves, tile = 128(M) x 64(N) x 32(K):
//   - Software pipeline: while computing WMMAs from LDS buffer `cur`, the
//     global loads for the next K-slab are in flight (registers), then
//     converted f32->f16 and stored into buffer `cur^1`.  One barrier per
//     iteration: it separates iteration j-1's reads of buffer (j+1)%2 from
//     iteration j's writes to the same buffer.
//   - All 6 staging loads per thread are issued before any wait (no per-load
//     s_wait_loadcnt serialization).
//   - Waves arranged 2(M) x 4(N); each wave owns 64x16 of C = 4 accumulators,
//     reusing one B fragment across 4 WMMAs per K-step.
//   - LDS row stride padded to 40 halves (80B): bank-conflict-free b128 reads,
//     16B alignment preserved.
//
// Fragment layout (CDNA5 ISA 7.12.2, 16-bit A 16x32):
//   lane l: row/col = l%16, kb = (l/16)*8
//   elements 0..7  -> K = kb+0 .. kb+7
//   elements 8..15 -> K = kb+16 .. kb+23
// C/D: vgpr r -> row = r + (l/16)*8, col = l%16.
// ---------------------------------------------------------------------------
#define BM 128
#define BN 64
#define BK 32
#define LDSS 40   // padded LDS row stride in halves

template<int ACT, bool HAS_BIAS>
__global__ __launch_bounds__(256)
void wmma_gemm_kernel(const float* __restrict__ A, const float* __restrict__ Wt,
                      const float* __restrict__ bias, float* __restrict__ C,
                      int K, int lda, int ldb, int ldc)
{
    __shared__ __align__(16) _Float16 sA[2][BM * LDSS];
    __shared__ __align__(16) _Float16 sB[2][BN * LDSS];

    const int t    = threadIdx.x;
    const int lane = t & 31;
    const int wave = t >> 5;
    const int wN   = wave & 3;       // 4 waves across N
    const int wM   = wave >> 2;      // 2 waves across M
    const int half = lane >> 4;      // 0: lanes 0-15, 1: lanes 16-31
    const int sub  = lane & 15;

    const int tileM = blockIdx.x * BM;
    const int tileN = blockIdx.y * BN;

    // ---- per-thread staging slots: A = 4 float4 (128x32), B = 2 float4 (64x32)
    const float* gA[4];
    const float* gB[2];
    int oA[4], oB[2];
    #pragma unroll
    for (int i = 0; i < 4; ++i) {
        const int q   = i * 256 + t;          // 1024 float4 slots
        const int row = q >> 3;               // 8 float4 per 32-wide row
        const int cq  = (q & 7) * 4;
        gA[i] = A + (size_t)(tileM + row) * lda + cq;
        oA[i] = row * LDSS + cq;
    }
    #pragma unroll
    for (int i = 0; i < 2; ++i) {
        const int q   = i * 256 + t;          // 512 float4 slots
        const int row = q >> 3;
        const int cq  = (q & 7) * 4;
        gB[i] = Wt + (size_t)(tileN + row) * ldb + cq;
        oB[i] = row * LDSS + cq;
    }

    float4 ra[4], rb[2];

    // ---- prologue: load + stage K-slab 0 into buffer 0
    #pragma unroll
    for (int i = 0; i < 4; ++i) ra[i] = *(const float4*)(gA[i]);
    #pragma unroll
    for (int i = 0; i < 2; ++i) rb[i] = *(const float4*)(gB[i]);
    #pragma unroll
    for (int i = 0; i < 4; ++i) {
        v4h h = { (_Float16)ra[i].x, (_Float16)ra[i].y,
                  (_Float16)ra[i].z, (_Float16)ra[i].w };
        *(v4h*)(sA[0] + oA[i]) = h;
    }
    #pragma unroll
    for (int i = 0; i < 2; ++i) {
        v4h h = { (_Float16)rb[i].x, (_Float16)rb[i].y,
                  (_Float16)rb[i].z, (_Float16)rb[i].w };
        *(v4h*)(sB[0] + oB[i]) = h;
    }

    v8f acc[4] = {};
    int cur = 0;

    for (int k0 = 0; k0 < K; k0 += BK) {
        __syncthreads();

        const bool hasNext = (k0 + BK) < K;
        if (hasNext) {
            // issue next-slab loads (kept in flight during compute)
            #pragma unroll
            for (int i = 0; i < 4; ++i) ra[i] = *(const float4*)(gA[i] + k0 + BK);
            #pragma unroll
            for (int i = 0; i < 2; ++i) rb[i] = *(const float4*)(gB[i] + k0 + BK);
            if (k0 + 2 * BK < K) {   // prefetch the slab after that
                __builtin_prefetch(gA[0] + k0 + 2 * BK, 0, 0);
                __builtin_prefetch(gB[0] + k0 + 2 * BK, 0, 0);
            }
        }

        // ---- compute from buffer `cur`
        const _Float16* __restrict__ bA = sA[cur];
        const _Float16* __restrict__ bB = sB[cur];
        v16h bf;
        {
            const int bro = (wN * 16 + sub) * LDSS + half * 8;
            v8h lo = *(const v8h*)(bB + bro);
            v8h hi = *(const v8h*)(bB + bro + 16);
            #pragma unroll
            for (int e = 0; e < 8; ++e) { bf[e] = lo[e]; bf[e + 8] = hi[e]; }
        }
        #pragma unroll
        for (int mt = 0; mt < 4; ++mt) {
            const int aro = (wM * 64 + mt * 16 + sub) * LDSS + half * 8;
            v16h af;
            v8h lo = *(const v8h*)(bA + aro);
            v8h hi = *(const v8h*)(bA + aro + 16);
            #pragma unroll
            for (int e = 0; e < 8; ++e) { af[e] = lo[e]; af[e + 8] = hi[e]; }
            acc[mt] = __builtin_amdgcn_wmma_f32_16x16x32_f16(
                /*neg_a=*/false, af, /*neg_b=*/false, bf,
                /*c_mod=*/(short)0, acc[mt], /*reuse_a=*/false, /*reuse_b=*/false);
        }

        // ---- stage next slab into the other buffer
        if (hasNext) {
            const int nxt = cur ^ 1;
            #pragma unroll
            for (int i = 0; i < 4; ++i) {
                v4h h = { (_Float16)ra[i].x, (_Float16)ra[i].y,
                          (_Float16)ra[i].z, (_Float16)ra[i].w };
                *(v4h*)(sA[nxt] + oA[i]) = h;
            }
            #pragma unroll
            for (int i = 0; i < 2; ++i) {
                v4h h = { (_Float16)rb[i].x, (_Float16)rb[i].y,
                          (_Float16)rb[i].z, (_Float16)rb[i].w };
                *(v4h*)(sB[nxt] + oB[i]) = h;
            }
        }
        cur ^= 1;
    }

    const int col = tileN + wN * 16 + sub;
    float bv = 0.0f;
    if (HAS_BIAS) bv = bias[col];

    #pragma unroll
    for (int mt = 0; mt < 4; ++mt) {
        #pragma unroll
        for (int r = 0; r < 8; ++r) {
            const int m = tileM + wM * 64 + mt * 16 + half * 8 + r;
            float v = acc[mt][r];
            if (HAS_BIAS) v += bv;
            if (ACT == 1)  v = softplus_f(v);
            C[(size_t)m * ldc + col] = v;
        }
    }
}

// ---------------------------------------------------------------------------
// Depthwise causal conv1d (k=4) + SiLU.  Input x = xz[:, :D_INNER] (row stride
// 2*D_INNER).  rev=1 runs on the time-reversed sequence; output stays in the
// (reversed) local time order, the gate-combine kernel un-reverses.
// ---------------------------------------------------------------------------
__global__ __launch_bounds__(256)
void conv_silu_kernel(const float* __restrict__ xz, const float* __restrict__ w,
                      const float* __restrict__ bias, float* __restrict__ out,
                      int rev)
{
    const int idx = blockIdx.x * 256 + threadIdx.x;       // over M * D_INNER
    if (idx >= kM * kDINNER) return;
    const int d   = idx & (kDINNER - 1);
    const int row = idx >> 10;
    const int l   = row & (kL - 1);
    const int bb  = row >> 10;

    float acc = bias[d];
    #pragma unroll
    for (int j = 0; j < 4; ++j) {
        const int pos = l - 3 + j;
        if (pos >= 0) {
            const int st = rev ? (kL - 1 - pos) : pos;
            acc += w[d * 4 + j] *
                   xz[((size_t)(bb * kL + st)) * (2 * kDINNER) + d];
        }
    }
    out[idx] = silu_f(acc);
}

// ---------------------------------------------------------------------------
// Selective scan.  One thread per (batch, channel): 4096 threads = 128 waves.
// Within a wave, B/C reads are wave-uniform (scalar broadcast), u/dt coalesced.
// ---------------------------------------------------------------------------
__global__ __launch_bounds__(256)
void selective_scan_kernel(const float* __restrict__ u,     // (M, D_INNER)
                           const float* __restrict__ dt,    // (M, D_INNER)
                           const float* __restrict__ proj,  // (M, 64): [32..48)=B, [48..64)=C
                           const float* __restrict__ A_log, // (D_INNER, N)
                           const float* __restrict__ Dskip, // (D_INNER)
                           float* __restrict__ y)           // (M, D_INNER)
{
    const int tid = blockIdx.x * 256 + threadIdx.x;        // 0 .. 4095
    const int d   = tid & (kDINNER - 1);
    const int bb  = tid >> 10;

    float Ac[kDSTATE];
    #pragma unroll
    for (int n = 0; n < kDSTATE; ++n) Ac[n] = -__expf(A_log[d * kDSTATE + n]);
    const float Dv = Dskip[d];

    float h[kDSTATE];
    #pragma unroll
    for (int n = 0; n < kDSTATE; ++n) h[n] = 0.0f;

    for (int l = 0; l < kL; ++l) {
        const size_t row = (size_t)bb * kL + l;
        const float dtv = dt[row * kDINNER + d];
        const float uv  = u[row * kDINNER + d];
        const float du  = dtv * uv;
        const float* __restrict__ pr = proj + row * 64;

        float acc = 0.0f;
        #pragma unroll
        for (int n = 0; n < kDSTATE; ++n) {
            const float dA = __expf(dtv * Ac[n]);
            h[n] = dA * h[n] + du * pr[32 + n];
            acc += h[n] * pr[48 + n];
        }
        y[row * kDINNER + d] = acc + uv * Dv;
    }
}

// ---------------------------------------------------------------------------
// y = (y_fwd + reverse_time(y_bwd)) * silu(z),  z = xz[:, D_INNER:]
// ---------------------------------------------------------------------------
__global__ __launch_bounds__(256)
void gate_combine_kernel(const float* __restrict__ yf, const float* __restrict__ yb,
                         const float* __restrict__ xz, float* __restrict__ out)
{
    const int idx = blockIdx.x * 256 + threadIdx.x;
    if (idx >= kM * kDINNER) return;
    const int d   = idx & (kDINNER - 1);
    const int row = idx >> 10;
    const int l   = row & (kL - 1);
    const int bb  = row >> 10;
    const size_t rrev = (size_t)(bb * kL + (kL - 1 - l));

    const float z = xz[(size_t)row * (2 * kDINNER) + kDINNER + d];
    out[idx] = (yf[idx] + yb[rrev * kDINNER + d]) * silu_f(z);
}

// ---------------------------------------------------------------------------
// out = residual + LayerNorm(x) * w + b    (one 256-thread block per row of 512)
// ---------------------------------------------------------------------------
__global__ __launch_bounds__(256)
void ln_residual_kernel(const float* __restrict__ x, const float* __restrict__ res,
                        const float* __restrict__ w, const float* __restrict__ b,
                        float* __restrict__ out)
{
    __shared__ float s1[256], s2[256];
    const int row = blockIdx.x;
    const int t   = threadIdx.x;
    const float* __restrict__ xr = x + (size_t)row * kDMODEL;

    const float v0 = xr[t], v1 = xr[t + 256];
    s1[t] = v0 + v1;
    s2[t] = v0 * v0 + v1 * v1;
    __syncthreads();
    for (int s = 128; s > 0; s >>= 1) {
        if (t < s) { s1[t] += s1[t + s]; s2[t] += s2[t + s]; }
        __syncthreads();
    }
    const float mean = s1[0] * (1.0f / kDMODEL);
    const float var  = s2[0] * (1.0f / kDMODEL) - mean * mean;
    const float rstd = rsqrtf(var + 1e-6f);

    const size_t base = (size_t)row * kDMODEL;
    #pragma unroll
    for (int p = 0; p < 2; ++p) {
        const int c = t + p * 256;
        out[base + c] = res[base + c] + (xr[c] - mean) * rstd * w[c] + b[c];
    }
}

__global__ __launch_bounds__(256)
void average_kernel(const float* __restrict__ a, const float* __restrict__ b,
                    float* __restrict__ out)
{
    const int idx = blockIdx.x * 256 + threadIdx.x;
    if (idx >= kM * kDMODEL) return;
    out[idx] = 0.5f * (a[idx] + b[idx]);
}

// ---------------------------------------------------------------------------
// Host-side orchestration
// ---------------------------------------------------------------------------
namespace {

struct Params {
    const float *in_proj_w, *conv_w, *conv_b, *xproj_w, *dtproj_w, *dtproj_b;
    const float *A_log, *Dskip, *out_proj_w, *ln_w, *ln_b;
};

struct Work {
    float *xz;            // (M, 2048)
    float *xc[2];         // (M, 1024) fwd/bwd conv output
    float *proj[2];       // (M, 64)
    float *dt[2];         // (M, 1024)
    float *y[2];          // (M, 1024)
    float *ycomb;         // (M, 1024)
    float *bm;            // (M, 512)
    float *t_at, *t_vt;   // (M, 512)
};

inline void run_gemm(const float* A, const float* W, const float* bias, float* C,
                     int M, int N, int K, int lda, int ldb, int ldc,
                     int act, hipStream_t s)
{
    dim3 grid(M / BM, N / BN), block(256);
    if (act == 1)
        wmma_gemm_kernel<1, true><<<grid, block, 0, s>>>(A, W, bias, C, K, lda, ldb, ldc);
    else if (bias)
        wmma_gemm_kernel<0, true><<<grid, block, 0, s>>>(A, W, bias, C, K, lda, ldb, ldc);
    else
        wmma_gemm_kernel<0, false><<<grid, block, 0, s>>>(A, W, nullptr, C, K, lda, ldb, ldc);
}

// One bidirectional Mamba block: state (M,512) -> bm (M,512)
void bimamba(const float* state, int li, const Params& P, const Work& W,
             hipStream_t s)
{
    const int nElemInner = kM * kDINNER;
    const int gridInner  = (nElemInner + 255) / 256;

    // in_proj: (M,512) x (2048,512)^T -> (M,2048)
    run_gemm(state, P.in_proj_w + (size_t)li * 2 * kDINNER * kDMODEL, nullptr,
             W.xz, kM, 2 * kDINNER, kDMODEL, kDMODEL, kDMODEL, 2 * kDINNER, 0, s);

    for (int dir = 0; dir < 2; ++dir) {
        const size_t pd = (size_t)(li * 2 + dir);
        // causal conv + SiLU on x (fwd) / reversed x (bwd)
        conv_silu_kernel<<<gridInner, 256, 0, s>>>(
            W.xz, P.conv_w + pd * kDINNER * 4, P.conv_b + pd * kDINNER,
            W.xc[dir], dir);

        // x_proj: (M,1024) x (64,1024)^T -> (M,64)  [dt_r | B | C]
        run_gemm(W.xc[dir], P.xproj_w + pd * (kDTRANK + 2 * kDSTATE) * kDINNER,
                 nullptr, W.proj[dir],
                 kM, kDTRANK + 2 * kDSTATE, kDINNER, kDINNER, kDINNER, 64, 0, s);

        // dt_proj + bias + softplus: (M,32) x (1024,32)^T -> (M,1024)
        run_gemm(W.proj[dir], P.dtproj_w + pd * kDINNER * kDTRANK,
                 P.dtproj_b + pd * kDINNER, W.dt[dir],
                 kM, kDINNER, kDTRANK, 64, kDTRANK, kDINNER, 1, s);

        // selective scan: exactly B*D_INNER = 4096 threads
        selective_scan_kernel<<<(kB * kDINNER) / 256, 256, 0, s>>>(
            W.xc[dir], W.dt[dir], W.proj[dir],
            P.A_log + pd * kDINNER * kDSTATE, P.Dskip + pd * kDINNER, W.y[dir]);
    }

    // (yf + flip(yb)) * silu(z)
    gate_combine_kernel<<<gridInner, 256, 0, s>>>(W.y[0], W.y[1], W.xz, W.ycomb);

    // out_proj: (M,1024) x (512,1024)^T -> (M,512)
    run_gemm(W.ycomb, P.out_proj_w + (size_t)li * kDMODEL * kDINNER, nullptr,
             W.bm, kM, kDMODEL, kDINNER, kDINNER, kDINNER, kDMODEL, 0, s);
}

inline void ln_res(const Params& P, int li, int slot, const float* res,
                   const float* bm, float* out, hipStream_t s)
{
    ln_residual_kernel<<<kM, 256, 0, s>>>(
        bm, res, P.ln_w + (size_t)(li * 2 + slot) * kDMODEL,
        P.ln_b + (size_t)(li * 2 + slot) * kDMODEL, out);
}

} // namespace

extern "C" void kernel_launch(void* const* d_in, const int* in_sizes, int n_in,
                              void* d_out, int out_size, void* d_ws, size_t ws_size,
                              hipStream_t stream)
{
    const float* a_x = (const float*)d_in[0];
    const float* v_x = (const float*)d_in[1];
    const float* t_x = (const float*)d_in[2];

    Params P;
    P.in_proj_w  = (const float*)d_in[3];
    P.conv_w     = (const float*)d_in[4];
    P.conv_b     = (const float*)d_in[5];
    P.xproj_w    = (const float*)d_in[6];
    P.dtproj_w   = (const float*)d_in[7];
    P.dtproj_b   = (const float*)d_in[8];
    P.A_log      = (const float*)d_in[9];
    P.Dskip      = (const float*)d_in[10];
    P.out_proj_w = (const float*)d_in[11];
    P.ln_w       = (const float*)d_in[12];
    P.ln_b       = (const float*)d_in[13];

    float* out  = (float*)d_out;
    const size_t seg = (size_t)kM * kDMODEL;
    float* a_st = out;            // running states live in d_out segments
    float* v_st = out + seg;
    float* t_st = out + 2 * seg;

    hipMemcpyAsync(a_st, a_x, seg * sizeof(float), hipMemcpyDeviceToDevice, stream);
    hipMemcpyAsync(v_st, v_x, seg * sizeof(float), hipMemcpyDeviceToDevice, stream);
    hipMemcpyAsync(t_st, t_x, seg * sizeof(float), hipMemcpyDeviceToDevice, stream);

    // Workspace layout
    float* w = (float*)d_ws;
    Work W;
    size_t off = 0;
    W.xz      = w + off; off += (size_t)kM * 2 * kDINNER;
    W.xc[0]   = w + off; off += (size_t)kM * kDINNER;
    W.xc[1]   = w + off; off += (size_t)kM * kDINNER;
    W.proj[0] = w + off; off += (size_t)kM * 64;
    W.proj[1] = w + off; off += (size_t)kM * 64;
    W.dt[0]   = w + off; off += (size_t)kM * kDINNER;
    W.dt[1]   = w + off; off += (size_t)kM * kDINNER;
    W.y[0]    = w + off; off += (size_t)kM * kDINNER;
    W.y[1]    = w + off; off += (size_t)kM * kDINNER;
    W.ycomb   = w + off; off += (size_t)kM * kDINNER;
    W.bm      = w + off; off += (size_t)kM * kDMODEL;
    W.t_at    = w + off; off += (size_t)kM * kDMODEL;
    W.t_vt    = w + off; off += (size_t)kM * kDMODEL;

    const int gridSeg = (int)((seg + 255) / 256);

    for (int i = 0; i < 2; ++i) {
        const int li_at = 2 * i, li_vt = 2 * i + 1;

        // _enc_layer(a, t, 2i)
        bimamba(a_st, li_at, P, W, stream);
        ln_res(P, li_at, 0, a_st, W.bm, a_st, stream);
        bimamba(t_st, li_at, P, W, stream);
        ln_res(P, li_at, 1, t_st, W.bm, W.t_at, stream);

        // _enc_layer(v, t, 2i+1)  (uses the same incoming t)
        bimamba(v_st, li_vt, P, W, stream);
        ln_res(P, li_vt, 0, v_st, W.bm, v_st, stream);
        bimamba(t_st, li_vt, P, W, stream);
        ln_res(P, li_vt, 1, t_st, W.bm, W.t_vt, stream);

        // t = (t_at + t_vt) / 2
        average_kernel<<<gridSeg, 256, 0, stream>>>(W.t_at, W.t_vt, t_st);
    }
}
